// HierarchicalMoE_45964740001797
// MI455X (gfx1250) — matrix operand order repaired and checked by reading, not compiled, not run on hardware
//
#include <hip/hip_runtime.h>
#include <hip/hip_bf16.h>
#include <stdint.h>

// ---------------------------------------------------------------------------
// Problem constants (HierarchicalMoE)
// ---------------------------------------------------------------------------
#define BATCH   2048
#define DIN     1024
#define DHID    2048
#define DOUT    1024
#define NHIGH   8
#define MLOW    4
#define NEXPERT (NHIGH * MLOW)   // 32
#define EGROUP  8                // experts per fused launch (z dim)

typedef __bf16 bf16_t;
typedef __attribute__((ext_vector_type(16))) __bf16 v16bf;
typedef __attribute__((ext_vector_type(8)))  __bf16 v8bf;
typedef __attribute__((ext_vector_type(4)))  __bf16 v4bf;
typedef __attribute__((ext_vector_type(8)))  float  v8f;
typedef __attribute__((ext_vector_type(4)))  int    v4i;

#if defined(__gfx1250__) && __has_builtin(__builtin_amdgcn_global_load_async_to_lds_b128)
#define HAS_ASYNC 1
#else
#define HAS_ASYNC 0
#endif

// 16B global -> LDS copy; async path (tracked by ASYNCcnt) when available.
// Builtin signature (probe-verified): (AS1 v4i*, AS3 v4i*, imm offset, cpol)
__device__ __forceinline__ void cp16_g2l(const bf16_t* g, bf16_t* l) {
#if HAS_ASYNC
    __builtin_amdgcn_global_load_async_to_lds_b128(
        (__attribute__((address_space(1))) v4i*)(uintptr_t)g,
        (__attribute__((address_space(3))) v4i*)(uintptr_t)l,
        0, 0);
#else
    *(uint4*)l = *(const uint4*)g;
#endif
}

__device__ __forceinline__ void wait_async_copies() {
#if HAS_ASYNC
#if __has_builtin(__builtin_amdgcn_s_wait_asynccnt)
    __builtin_amdgcn_s_wait_asynccnt(0);
#else
    asm volatile("s_wait_asynccnt 0" ::: "memory");
#endif
#endif
}

union Frag {
    v16bf v;
    v8bf  h[2];
};

__device__ __forceinline__ v8f wmma_bf16(const Frag& a, const Frag& b, v8f c) {
#if defined(__gfx1250__)
    // D(f32 16x16) = A(bf16 16x32) * B(bf16 32x16) + C
    return __builtin_amdgcn_wmma_f32_16x16x32_bf16(
        false, a.v, false, b.v, (short)0, c, false, false);
#else
    (void)a; (void)b;
    return c;
#endif
}

// ---------------------------------------------------------------------------
// x f32 -> bf16 conversion
// ---------------------------------------------------------------------------
__global__ __launch_bounds__(256)
void cvt_kernel(const float* __restrict__ in, bf16_t* __restrict__ out, int n) {
    int i = (blockIdx.x * 256 + threadIdx.x) * 4;
    if (i + 3 < n) {
        float4 f = *(const float4*)(in + i);
        v4bf v;
        v[0] = (bf16_t)f.x; v[1] = (bf16_t)f.y;
        v[2] = (bf16_t)f.z; v[3] = (bf16_t)f.w;
        *(v4bf*)(out + i) = v;
    }
}

// ---------------------------------------------------------------------------
// Gating: one block per batch row. 40 length-1024 dots + softmaxes.
// ---------------------------------------------------------------------------
__global__ __launch_bounds__(256)
void gate_kernel(const float* __restrict__ x,
                 const float* __restrict__ Wh, const float* __restrict__ bh,
                 const float* __restrict__ Wg, const float* __restrict__ bg,
                 float* __restrict__ outHigh, float* __restrict__ outLow,
                 float* __restrict__ outComb, float* __restrict__ combWs) {
    const int b = blockIdx.x;
    const int t = threadIdx.x;
    const int lane = t & 31;
    const int wv = t >> 5;

    float ph[NHIGH];
    float pg[NEXPERT];
#pragma unroll
    for (int n = 0; n < NHIGH; ++n) ph[n] = 0.f;
#pragma unroll
    for (int o = 0; o < NEXPERT; ++o) pg[o] = 0.f;

    for (int i = 0; i < DIN / 256; ++i) {
        const int e = t + i * 256;
        const float xv = x[(size_t)b * DIN + e];
#pragma unroll
        for (int n = 0; n < NHIGH; ++n) ph[n] += xv * Wh[(size_t)e * NHIGH + n];
#pragma unroll
        for (int n = 0; n < NHIGH; ++n)
#pragma unroll
            for (int m = 0; m < MLOW; ++m)
                pg[n * MLOW + m] += xv * Wg[((size_t)n * DIN + e) * MLOW + m];
    }
    // wave32 reduction
#pragma unroll
    for (int n = 0; n < NHIGH; ++n)
        for (int s = 16; s > 0; s >>= 1) ph[n] += __shfl_xor(ph[n], s, 32);
#pragma unroll
    for (int o = 0; o < NEXPERT; ++o)
        for (int s = 16; s > 0; s >>= 1) pg[o] += __shfl_xor(pg[o], s, 32);

    __shared__ float red[NHIGH + NEXPERT][8];
    __shared__ float fin[NHIGH + NEXPERT];
    if (lane == 0) {
#pragma unroll
        for (int n = 0; n < NHIGH; ++n) red[n][wv] = ph[n];
#pragma unroll
        for (int o = 0; o < NEXPERT; ++o) red[NHIGH + o][wv] = pg[o];
    }
    __syncthreads();
    if (t < NHIGH + NEXPERT) {
        float s = 0.f;
#pragma unroll
        for (int w = 0; w < 8; ++w) s += red[t][w];
        fin[t] = s;
    }
    __syncthreads();
    if (t == 0) {
        float hi[NHIGH];
        float mx = -3.4e38f;
#pragma unroll
        for (int n = 0; n < NHIGH; ++n) { hi[n] = fin[n] + bh[n]; mx = fmaxf(mx, hi[n]); }
        float se = 0.f;
#pragma unroll
        for (int n = 0; n < NHIGH; ++n) { hi[n] = __expf(hi[n] - mx); se += hi[n]; }
        float inv = 1.f / se;
#pragma unroll
        for (int n = 0; n < NHIGH; ++n) {
            hi[n] *= inv;
            outHigh[(size_t)b * NHIGH + n] = hi[n];
        }
#pragma unroll
        for (int n = 0; n < NHIGH; ++n) {
            float lo[MLOW];
            float m2 = -3.4e38f;
#pragma unroll
            for (int m = 0; m < MLOW; ++m) {
                lo[m] = fin[NHIGH + n * MLOW + m] + bg[n * MLOW + m];
                m2 = fmaxf(m2, lo[m]);
            }
            float s2 = 0.f;
#pragma unroll
            for (int m = 0; m < MLOW; ++m) { lo[m] = __expf(lo[m] - m2); s2 += lo[m]; }
            float i2 = 1.f / s2;
#pragma unroll
            for (int m = 0; m < MLOW; ++m) {
                float lp = lo[m] * i2;
                float cb = hi[n] * lp;
                size_t idx = (size_t)b * NEXPERT + n * MLOW + m;
                outLow[idx]  = lp;
                outComb[idx] = cb;
                combWs[idx]  = cb;
            }
        }
    }
}

// ---------------------------------------------------------------------------
// Fused expert GEMM (dims as template params so the K-loop address math is
// loop-invariant / strength-reduced; no 64-bit multiplies in the hot loop).
// MODE 1:  out(bf16) = relu(A(bf16) @ W(f32->bf16) + bias)       [layer 1]
// MODE 2:  d_out(f32) atomically += gate * (A @ W + bias)        [layer 2]
// A row-major [rows x K] bf16, W row-major [K x N] f32.
// Block: 256 threads (8 wave32), tile 128x256, K-step 32, double-buffered LDS.
// Wave tile 64x64 -> 16 accumulators, A/B fragments each reused 4x.
// ---------------------------------------------------------------------------
template <int MODE, int K, int LDA, int LDB, int LDO>
__global__ __launch_bounds__(256)
void moe_gemm(const bf16_t* __restrict__ Abase, size_t aZ,
              const float* __restrict__ Wbase, size_t wZ,
              const float* __restrict__ biasBase, int biasZ,
              void* __restrict__ outBase, size_t oZ,
              const float* __restrict__ comb, int e0) {
    constexpr int BM = 128, BN = 256, BK = 32, PAD = 40;  // PAD: 16B align + bank spread
    __shared__ bf16_t sA[2][BM * PAD];  // A tile, row-major [m][k]     (20 KB)
    __shared__ bf16_t sB[2][BN * PAD];  // W tile, transposed [n][k]    (40 KB)
    __shared__ float  sBias[BN];
    __shared__ float  sGate[BM];

    const int tid  = threadIdx.x;
    const int lane = tid & 31;
    const int wave = tid >> 5;
    const int wm   = wave & 1;   // 2 wave-rows  (64 rows each)
    const int wn   = wave >> 1;  // 4 wave-cols  (64 cols each)
    const int z    = blockIdx.z;
    const int row0 = blockIdx.y * BM;
    const int col0 = blockIdx.x * BN;

    const bf16_t* A    = Abase + (size_t)z * aZ;
    const float*  W    = Wbase + (size_t)z * wZ;
    const float*  bias = biasBase + (size_t)z * biasZ;

    sBias[tid] = bias[col0 + tid];
    if (MODE == 2 && tid < BM)
        sGate[tid] = comb[(size_t)(row0 + tid) * NEXPERT + (e0 + z)];

    auto loadA = [&](int s, int kk) {
#pragma unroll
        for (int i = 0; i < 2; ++i) {
            int c  = tid * 2 + i;        // 512 chunks of 8 bf16
            int r  = c >> 2;
            int kc = (c & 3) * 8;
            cp16_g2l(A + (size_t)(row0 + r) * LDA + kk + kc,
                     &sA[s][r * PAD + kc]);
        }
    };
    auto loadB = [&](int s, int kk) {
#pragma unroll
        for (int i = 0; i < 8; ++i) {
            int c  = tid + i * 256;      // 2048 chunks: [n][4 k-values]
            int n  = c & 255;
            int k4 = (c >> 8) * 4;
            float f0 = W[(size_t)(kk + k4 + 0) * LDB + col0 + n];
            float f1 = W[(size_t)(kk + k4 + 1) * LDB + col0 + n];
            float f2 = W[(size_t)(kk + k4 + 2) * LDB + col0 + n];
            float f3 = W[(size_t)(kk + k4 + 3) * LDB + col0 + n];
            v4bf v;
            v[0] = (bf16_t)f0; v[1] = (bf16_t)f1;
            v[2] = (bf16_t)f2; v[3] = (bf16_t)f3;
            *(v4bf*)&sB[s][n * PAD + k4] = v;   // 8B ds_store, transposed
        }
    };

    v8f acc[4][4];
#pragma unroll
    for (int i = 0; i < 4; ++i)
#pragma unroll
        for (int j = 0; j < 4; ++j) acc[i][j] = 0.f;

    constexpr int ksteps = K / BK;
    loadA(0, 0);
    loadB(0, 0);
    wait_async_copies();
    __syncthreads();

    for (int ks = 0; ks < ksteps; ++ks) {
        const int cur = ks & 1;
        if (ks + 1 < ksteps) {  // prefetch next stage while computing
            loadA(cur ^ 1, (ks + 1) * BK);
            loadB(cur ^ 1, (ks + 1) * BK);
        }
        Frag a[4], b[4];
#pragma unroll
        for (int i = 0; i < 4; ++i) {
            // ISA 16-bit A layout: lanes0-15 K0-7/16-23, lanes16-31 K8-15/24-31
            int r   = wm * 64 + i * 16 + (lane & 15);
            int off = r * PAD + (lane >> 4) * 8;
            a[i].h[0] = *(const v8bf*)&sA[cur][off];
            a[i].h[1] = *(const v8bf*)&sA[cur][off + 16];
        }
#pragma unroll
        for (int j = 0; j < 4; ++j) {
            // B lane holds column n, K order matches A: lanes0-15 K0-7/16-23, lanes16-31 K8-15/24-31 (transposed LDS)
            int n   = wn * 64 + j * 16 + (lane & 15);
            int off = n * PAD + (lane >> 4) * 8;
            b[j].h[0] = *(const v8bf*)&sB[cur][off];
            b[j].h[1] = *(const v8bf*)&sB[cur][off + 16];
        }
#pragma unroll
        for (int i = 0; i < 4; ++i)
#pragma unroll
            for (int j = 0; j < 4; ++j)
                acc[i][j] = wmma_bf16(a[i], b[j], acc[i][j]);
        wait_async_copies();
        __syncthreads();
    }

    // C/D layout: lanes0-15 -> N=lane, M=v; lanes16-31 -> N=lane-16, M=v+8
    const int half8 = (lane >> 4) * 8;
    if (MODE == 1) {
        bf16_t* H = (bf16_t*)outBase + (size_t)z * oZ;
#pragma unroll
        for (int i = 0; i < 4; ++i) {
            const int rl = wm * 64 + i * 16 + half8;
#pragma unroll
            for (int j = 0; j < 4; ++j) {
                const int cl = wn * 64 + j * 16 + (lane & 15);
                const float bb = sBias[cl];
#pragma unroll
                for (int v = 0; v < 8; ++v) {
                    float val = acc[i][j][v] + bb;
                    val = val > 0.f ? val : 0.f;
                    H[(size_t)(row0 + rl + v) * LDO + col0 + cl] = (bf16_t)val;
                }
            }
        }
    } else {
        float* O = (float*)outBase;
#pragma unroll
        for (int i = 0; i < 4; ++i) {
            const int rl = wm * 64 + i * 16 + half8;
#pragma unroll
            for (int j = 0; j < 4; ++j) {
                const int cl = wn * 64 + j * 16 + (lane & 15);
                const float bb = sBias[cl];
#pragma unroll
                for (int v = 0; v < 8; ++v) {
                    const float g = sGate[rl + v];
                    atomicAdd(&O[(size_t)(row0 + rl + v) * LDO + col0 + cl],
                              g * (acc[i][j][v] + bb));
                }
            }
        }
    }
}

// ---------------------------------------------------------------------------
// Launch
// ---------------------------------------------------------------------------
extern "C" void kernel_launch(void* const* d_in, const int* in_sizes, int n_in,
                              void* d_out, int out_size, void* d_ws, size_t ws_size,
                              hipStream_t stream) {
    (void)in_sizes; (void)n_in; (void)out_size; (void)ws_size;
    const float* x  = (const float*)d_in[0];
    const float* Wh = (const float*)d_in[1];
    const float* bh = (const float*)d_in[2];
    const float* Wg = (const float*)d_in[3];
    const float* bg = (const float*)d_in[4];
    const float* W1 = (const float*)d_in[5];
    const float* b1 = (const float*)d_in[6];
    const float* W2 = (const float*)d_in[7];
    const float* b2 = (const float*)d_in[8];

    float* out     = (float*)d_out;                        // (B, DOUT)
    float* outHigh = out + (size_t)BATCH * DOUT;           // (B, N)
    float* outLow  = outHigh + (size_t)BATCH * NHIGH;      // (B, N, M)
    float* outComb = outLow + (size_t)BATCH * NEXPERT;     // (B, N*M)

    // Workspace layout (~69 MB): x_bf16 | combined | h_bf16[EGROUP]
    char* ws = (char*)d_ws;
    bf16_t* xbf     = (bf16_t*)ws;                                   // 4 MB
    float*  combWs  = (float*)(ws + (size_t)4 * 1024 * 1024);        // 256 KB
    bf16_t* h_ws    = (bf16_t*)(ws + (size_t)4 * 1024 * 1024 + 512 * 1024); // 64 MB

    (void)hipMemsetAsync(out, 0, (size_t)BATCH * DOUT * sizeof(float), stream);

    cvt_kernel<<<(BATCH * DIN) / (256 * 4), 256, 0, stream>>>(x, xbf, BATCH * DIN);
    gate_kernel<<<BATCH, 256, 0, stream>>>(x, Wh, bh, Wg, bg,
                                           outHigh, outLow, outComb, combWs);

    for (int g = 0; g < NEXPERT / EGROUP; ++g) {
        const int e0 = g * EGROUP;
        // layer 1: h[e] = relu(x @ W1[e] + b1[e])  -> bf16 ws
        moe_gemm<1, DIN, DIN, DHID, DHID>
            <<<dim3(DHID / 256, BATCH / 128, EGROUP), 256, 0, stream>>>(
                xbf, (size_t)0,
                W1 + (size_t)e0 * DIN * DHID, (size_t)DIN * DHID,
                b1 + (size_t)e0 * DHID, DHID,
                (void*)h_ws, (size_t)BATCH * DHID,
                nullptr, e0);
        // layer 2: out += comb[:,e] * (h[e] @ W2[e] + b2[e])
        moe_gemm<2, DHID, DHID, DOUT, DOUT>
            <<<dim3(DOUT / 256, BATCH / 128, EGROUP), 256, 0, stream>>>(
                h_ws, (size_t)BATCH * DHID,
                W2 + (size_t)e0 * DHID * DOUT, (size_t)DHID * DOUT,
                b2 + (size_t)e0 * DOUT, DOUT,
                (void*)out, (size_t)0,
                combWs, e0);
    }
}
